// RNN_decoder_model_8117488190097
// MI455X (gfx1250) — matrix operand order repaired and checked by reading, not compiled, not run on hardware
//
#include <hip/hip_runtime.h>
#include <hip/hip_bf16.h>

// ---------------------------------------------------------------------------
// Problem constants (fixed by the reference's setup_inputs)
// ---------------------------------------------------------------------------
#define HID    128
#define G4     512          // 4*HID
#define BATCH  256
#define LAYERS 4
#define VOCAB  32000
#define TSTEPS 32           // max_sentence_len
#define ROWS   32           // batch rows per LSTM workgroup (8 workgroups)

typedef __attribute__((ext_vector_type(16))) __bf16 v16bf;
typedef __attribute__((ext_vector_type(8)))  float  v8f;

union BF16x16 { v16bf v; uint4 q[2]; unsigned short s[16]; };

// gfx1250 async memory path (ASYNCcnt-tracked LDS DMA), guarded for safety.
// Builtin signature (per hipcc diagnostic): (v4i AS1*, v4i AS3*, imm off, imm cpol)
#if __has_builtin(__builtin_amdgcn_global_load_async_to_lds_b128) && \
    __has_builtin(__builtin_amdgcn_s_wait_asynccnt)
#define HAVE_ASYNC_LDS 1
typedef int v4i __attribute__((__vector_size__(16)));
typedef __attribute__((address_space(1))) v4i gv4i;
typedef __attribute__((address_space(3))) v4i lv4i;
#else
#define HAVE_ASYNC_LDS 0
#endif

__device__ __forceinline__ unsigned short f2bf(float f) {
    unsigned u = __float_as_uint(f);
    unsigned r = u + 0x7FFFu + ((u >> 16) & 1u);   // round-to-nearest-even
    return (unsigned short)(r >> 16);
}

__device__ __forceinline__ float sigf(float x) { return 1.0f / (1.0f + __expf(-x)); }

// A-fragment (16-bit A, 16x32, wave32): lane = hi*16+ln holds row (m0+ln),
// elements 0..15 = K[k+hi*16 .. +16) contiguous (matches B fragment).
__device__ __forceinline__ v16bf load_a_tile(const unsigned short* base, int row, int k, int hi) {
    const unsigned short* p = base + row * HID + k + hi * 16;
    BF16x16 u;
    u.q[0] = *(const uint4*)(p);
    u.q[1] = *(const uint4*)(p + 8);
    return u.v;
}

// B-fragment (16-bit B, 32x16): lane = hi*16+ln holds column n, K = k+hi*16 .. +16
// contiguous.  For X @ W.T with row-major W[out][K], that is W[n][k+hi*16 ...],
// i.e. 16 contiguous bf16 (32 bytes).
__device__ __forceinline__ v16bf load_b_tile(const unsigned short* p) {
    BF16x16 u;
    u.q[0] = *(const uint4*)(p);
    u.q[1] = *(const uint4*)(p + 8);
    return u.v;
}

#define WMMA_BF16(A, B, C) \
    __builtin_amdgcn_wmma_f32_16x16x32_bf16(false, (A), false, (B), (short)0, (C), false, false)

// ---------------------------------------------------------------------------
// f32 -> bf16 conversion (weight prep)
// ---------------------------------------------------------------------------
__global__ void f32_to_bf16_kernel(const float* __restrict__ src,
                                   unsigned short* __restrict__ dst, int n) {
    int i = blockIdx.x * blockDim.x + threadIdx.x;
    if (i < n) dst[i] = f2bf(src[i]);
}

// ---------------------------------------------------------------------------
// Fused LSTM step: embedding gather -> 4 LSTM layers -> re-projection.
// 8 workgroups x 256 threads (8 waves); each owns 32 disjoint batch rows.
// Gate accumulators (i,f,g,o) stay in registers; activations done lane-local
// per the C-matrix layout (VGPR r, lane l -> M = r + 8*(l>=16), N = l&15).
// ---------------------------------------------------------------------------
__global__ __launch_bounds__(256) void lstm_step_kernel(
    int step, const int* __restrict__ prev, const int* __restrict__ eos,
    const float* __restrict__ W_in, const float* __restrict__ b_in,
    const unsigned short* __restrict__ Wih_bf, const unsigned short* __restrict__ Whh_bf,
    const float* __restrict__ b_ih, const float* __restrict__ b_hh,
    const unsigned short* __restrict__ Wre_bf, const float* __restrict__ b_re,
    float* __restrict__ hid, float* __restrict__ cel,
    unsigned short* __restrict__ emb_bf)
{
    __shared__ unsigned short x_s[ROWS][HID];   // layer input  (bf16)
    __shared__ unsigned short h_s[ROWS][HID];   // h_{t-1}[l]   (bf16)

    const int b0   = blockIdx.x * ROWS;
    const int t    = threadIdx.x;
    const int lane = t & 31;
    const int wave = t >> 5;                    // 8 waves
    const int hi   = lane >> 4;
    const int ln   = lane & 15;

    // --- embedding gather: x = W_in[prev] + b_in -----------------------------
    for (int i = t; i < ROWS * HID; i += 256) {
        int m = i >> 7, e = i & (HID - 1);
        int tok = (step == 0) ? eos[0] : prev[b0 + m];
        x_s[m][e] = f2bf(W_in[(size_t)tok * HID + e] + b_in[e]);
    }
    __syncthreads();

    v8f acc[2][4];
    const v8f vzero = {};

    for (int l = 0; l < LAYERS; ++l) {
        // load previous hidden state of this layer
        for (int i = t; i < ROWS * HID; i += 256) {
            int m = i >> 7, e = i & (HID - 1);
            h_s[m][e] = f2bf(hid[((size_t)l * BATCH + b0 + m) * HID + e]);
        }
        __syncthreads();

        const unsigned short* Wih_l = Wih_bf + (size_t)l * G4 * HID;
        const unsigned short* Whh_l = Whh_bf + (size_t)l * G4 * HID;

        // GEMM: 16 activation tiles (2 M x 8 U), wave handles 2; each needs
        // the i,f,g,o gate tiles => 4 WMMA per k-step, K = 128(x) + 128(h).
        for (int j = 0; j < 2; ++j) {
            int at = wave + 8 * j;              // 0..15
            int mt = at & 1;
            int u0 = (at >> 1) * 16;            // 0..112
            int arow = mt * 16 + ln;
            for (int g = 0; g < 4; ++g) acc[j][g] = vzero;
            for (int k = 0; k < HID; k += 32) {
                v16bf a = load_a_tile(&x_s[0][0], arow, k, hi);
                for (int g = 0; g < 4; ++g) {
                    int n = g * HID + u0 + ln;
                    v16bf bm = load_b_tile(Wih_l + (size_t)n * HID + k + hi * 16);
                    acc[j][g] = WMMA_BF16(a, bm, acc[j][g]);
                }
            }
            for (int k = 0; k < HID; k += 32) {
                v16bf a = load_a_tile(&h_s[0][0], arow, k, hi);
                for (int g = 0; g < 4; ++g) {
                    int n = g * HID + u0 + ln;
                    v16bf bm = load_b_tile(Whh_l + (size_t)n * HID + k + hi * 16);
                    acc[j][g] = WMMA_BF16(a, bm, acc[j][g]);
                }
            }
        }
        __syncthreads();                        // all x_s/h_s reads done

        // activations, state update, and write new x (=h) into LDS
        for (int j = 0; j < 2; ++j) {
            int at = wave + 8 * j;
            int mt = at & 1;
            int u0 = (at >> 1) * 16;
            int col = u0 + ln;
            float bi = b_ih[l * G4 + col]           + b_hh[l * G4 + col];
            float bf = b_ih[l * G4 + HID + col]     + b_hh[l * G4 + HID + col];
            float bg = b_ih[l * G4 + 2 * HID + col] + b_hh[l * G4 + 2 * HID + col];
            float bo = b_ih[l * G4 + 3 * HID + col] + b_hh[l * G4 + 3 * HID + col];
            for (int r = 0; r < 8; ++r) {
                int mrow = mt * 16 + r + 8 * hi;
                size_t ci = ((size_t)l * BATCH + b0 + mrow) * HID + col;
                float gi = acc[j][0][r] + bi;
                float gf = acc[j][1][r] + bf;
                float gg = acc[j][2][r] + bg;
                float go = acc[j][3][r] + bo;
                float c  = sigf(gf) * cel[ci] + sigf(gi) * tanhf(gg);
                float h  = sigf(go) * tanhf(c);
                cel[ci] = c;
                hid[ci] = h;
                x_s[mrow][col] = f2bf(h);
            }
        }
        __syncthreads();
    }

    // --- re-projection: emb = x @ W_re.T + b_re (2 M x 8 N tiles) ------------
    for (int j = 0; j < 2; ++j) {
        int at = wave + 8 * j;                  // 0..15
        int mt = at & 1;
        int u0 = (at >> 1) * 16;
        int arow = mt * 16 + ln;
        int n = u0 + ln;
        v8f a2 = vzero;
        for (int k = 0; k < HID; k += 32) {
            v16bf a  = load_a_tile(&x_s[0][0], arow, k, hi);
            v16bf bm = load_b_tile(Wre_bf + (size_t)n * HID + k + hi * 16);
            a2 = WMMA_BF16(a, bm, a2);
        }
        float bias = b_re[n];
        for (int r = 0; r < 8; ++r) {
            int mrow = mt * 16 + r + 8 * hi;
            emb_bf[(size_t)(b0 + mrow) * HID + n] = f2bf(a2[r] + bias);
        }
    }
}

// ---------------------------------------------------------------------------
// Vocab projection: logits = emb @ W_lo.T + b_lo   ([256 x 32000], K=128).
// 250 workgroups x 8 waves; each wave owns one 16-column vocab tile, keeps its
// B fragments (4 x v16bf) in registers, and loops over all 16 batch blocks,
// so W_lo (bf16, 8.2 MB, L2-resident) is streamed once per step.
// The shared emb operand (64 KB bf16) is staged once per workgroup into LDS
// via GLOBAL_LOAD_ASYNC_TO_LDS_B128 (ASYNCcnt), then read as ds_load_b128
// A-fragments -- avoids ~128 MB/step of redundant VMEM traffic.
// ---------------------------------------------------------------------------
__global__ __launch_bounds__(256) void logits_kernel(
    const unsigned short* __restrict__ emb_bf,
    const unsigned short* __restrict__ Wlo_bf,
    const float* __restrict__ b_lo,
    float* __restrict__ logits)
{
    __shared__ unsigned short emb_s[BATCH][HID];    // 64 KB

    const int t    = threadIdx.x;
    const int lane = t & 31;
    const int wave = t >> 5;
    const int hi = lane >> 4, ln = lane & 15;
    const int nt = blockIdx.x * 8 + wave;           // 0..1999
    const int n  = nt * 16 + ln;

    // ---- stage emb into LDS (4096 x 16B chunks, 16 per thread) -------------
    {
        const uint4* gsrc = (const uint4*)emb_bf;
        uint4*       ldst = (uint4*)&emb_s[0][0];
#if HAVE_ASYNC_LDS
        for (int i = t; i < (BATCH * HID * 2) / 16; i += 256) {
            __builtin_amdgcn_global_load_async_to_lds_b128(
                (gv4i*)(gsrc + i), (lv4i*)(ldst + i), 0, 0);
        }
        __builtin_amdgcn_s_wait_asynccnt(0);
#else
        for (int i = t; i < (BATCH * HID * 2) / 16; i += 256)
            ldst[i] = gsrc[i];
#endif
    }

    // preload the wave's B fragments for all 4 k-steps (register-resident)
    v16bf bm[4];
    for (int ks = 0; ks < 4; ++ks)
        bm[ks] = load_b_tile(Wlo_bf + (size_t)n * HID + ks * 32 + hi * 16);
    const float bias = b_lo[n];

    __syncthreads();

    for (int mb = 0; mb < BATCH / 16; ++mb) {
        v8f acc = {};
        int arow = mb * 16 + ln;
        for (int ks = 0; ks < 4; ++ks) {
            v16bf a = load_a_tile(&emb_s[0][0], arow, ks * 32, hi);
            acc = WMMA_BF16(a, bm[ks], acc);
        }
        for (int r = 0; r < 8; ++r) {
            int row = mb * 16 + r + 8 * hi;
            logits[(size_t)row * VOCAB + n] = acc[r] + bias;
        }
    }
}

// ---------------------------------------------------------------------------
// Per-row softmax + argmax + prob write.  One workgroup per batch row; the
// row's 128 KB of logits stays hot in L2 across the passes.  Pass 2 stores
// exp(x - max) back into the (privately owned) logits row so pass 3 is a
// pure scale+store instead of recomputing __expf over 32000 elements.
// ---------------------------------------------------------------------------
__global__ __launch_bounds__(256) void softmax_kernel(
    float* __restrict__ logits, float* __restrict__ out,
    int* __restrict__ prev)
{
    const int b = blockIdx.x;
    const int t = threadIdx.x;
    float* row = logits + (size_t)b * VOCAB;

    __shared__ float smax[256];
    __shared__ int   sidx[256];
    __shared__ float ssum[256];

    // pass 1: max (+ first-occurrence argmax, matching jnp.argmax)
    float mx = -3.402823e38f; int mi = VOCAB;
    for (int v = t; v < VOCAB; v += 256) {
        float x = row[v];
        if (x > mx) { mx = x; mi = v; }
    }
    smax[t] = mx; sidx[t] = mi;
    __syncthreads();
    for (int s = 128; s > 0; s >>= 1) {
        if (t < s) {
            float o = smax[t + s]; int oi = sidx[t + s];
            if (o > smax[t] || (o == smax[t] && oi < sidx[t])) { smax[t] = o; sidx[t] = oi; }
        }
        __syncthreads();
    }
    float rmax = smax[0];
    if (t == 0) prev[b] = sidx[0];

    // pass 2: sum of exp; store exp values in place
    float sum = 0.0f;
    for (int v = t; v < VOCAB; v += 256) {
        float e = __expf(row[v] - rmax);
        row[v] = e;
        sum += e;
    }
    ssum[t] = sum;
    __syncthreads();
    for (int s = 128; s > 0; s >>= 1) {
        if (t < s) ssum[t] += ssum[t + s];
        __syncthreads();
    }
    float inv = 1.0f / ssum[0];

    // pass 3: scale + write probabilities
    for (int v = t; v < VOCAB; v += 256)
        out[(size_t)b * VOCAB + v] = row[v] * inv;
}

// ---------------------------------------------------------------------------
// Host launcher
// ---------------------------------------------------------------------------
extern "C" void kernel_launch(void* const* d_in, const int* in_sizes, int n_in,
                              void* d_out, int out_size, void* d_ws, size_t ws_size,
                              hipStream_t stream) {
    (void)in_sizes; (void)n_in; (void)out_size; (void)ws_size;

    const float* hidden = (const float*)d_in[0];
    const float* cell   = (const float*)d_in[1];
    const float* W_in   = (const float*)d_in[2];
    const float* b_in   = (const float*)d_in[3];
    const float* W_ih   = (const float*)d_in[4];
    const float* W_hh   = (const float*)d_in[5];
    const float* b_ih   = (const float*)d_in[6];
    const float* b_hh   = (const float*)d_in[7];
    const float* W_re   = (const float*)d_in[8];
    const float* b_re   = (const float*)d_in[9];
    const float* W_lo   = (const float*)d_in[10];
    const float* b_lo   = (const float*)d_in[11];
    const int*   eos    = (const int*)d_in[12];
    float* out = (float*)d_out;

    // workspace carve-up (~43 MB)
    char* ws = (char*)d_ws;
    size_t off = 0;
    auto take = [&](size_t bytes) {
        char* p = ws + off;
        off += (bytes + 255) & ~(size_t)255;
        return p;
    };
    unsigned short* Wih_bf = (unsigned short*)take((size_t)LAYERS * G4 * HID * 2);
    unsigned short* Whh_bf = (unsigned short*)take((size_t)LAYERS * G4 * HID * 2);
    unsigned short* Wre_bf = (unsigned short*)take((size_t)HID * HID * 2);
    unsigned short* Wlo_bf = (unsigned short*)take((size_t)VOCAB * HID * 2);
    float*          hid_ws = (float*)take((size_t)LAYERS * BATCH * HID * 4);
    float*          cel_ws = (float*)take((size_t)LAYERS * BATCH * HID * 4);
    unsigned short* emb_bf = (unsigned short*)take((size_t)BATCH * HID * 2);
    float*          logits = (float*)take((size_t)BATCH * VOCAB * 4);
    int*            prev   = (int*)take((size_t)BATCH * 4);

    // weight prep: f32 -> bf16 (once per call; deterministic)
    {
        int n;
        n = LAYERS * G4 * HID;
        f32_to_bf16_kernel<<<(n + 255) / 256, 256, 0, stream>>>(W_ih, Wih_bf, n);
        f32_to_bf16_kernel<<<(n + 255) / 256, 256, 0, stream>>>(W_hh, Whh_bf, n);
        n = HID * HID;
        f32_to_bf16_kernel<<<(n + 255) / 256, 256, 0, stream>>>(W_re, Wre_bf, n);
        n = VOCAB * HID;
        f32_to_bf16_kernel<<<(n + 255) / 256, 256, 0, stream>>>(W_lo, Wlo_bf, n);
    }
    // state init (inputs must not be mutated)
    (void)hipMemcpyAsync(hid_ws, hidden, (size_t)LAYERS * BATCH * HID * 4,
                         hipMemcpyDeviceToDevice, stream);
    (void)hipMemcpyAsync(cel_ws, cell, (size_t)LAYERS * BATCH * HID * 4,
                         hipMemcpyDeviceToDevice, stream);

    // sequential decode: 32 steps (max_sentence_len fixed by setup_inputs)
    for (int s = 0; s < TSTEPS; ++s) {
        lstm_step_kernel<<<BATCH / ROWS, 256, 0, stream>>>(
            s, prev, eos, W_in, b_in, Wih_bf, Whh_bf, b_ih, b_hh,
            Wre_bf, b_re, hid_ws, cel_ws, emb_bf);
        logits_kernel<<<VOCAB / (16 * 8), 256, 0, stream>>>(
            emb_bf, Wlo_bf, b_lo, logits);
        softmax_kernel<<<BATCH, 256, 0, stream>>>(
            logits, out + (size_t)s * BATCH * VOCAB, prev);
    }
}
